// MultiHeadAttention_3624952398459
// MI455X (gfx1250) — matrix operand order repaired and checked by reading, not compiled, not run on hardware
//
#include <hip/hip_runtime.h>
#include <hip/hip_bf16.h>

// ---------------------------------------------------------------------------
// MI455X (gfx1250) fused MHA: QKV GEMM -> RoPE -> flash-attention -> out GEMM
// WMMA bf16 core + async-to-LDS (ASYNCcnt) streaming for attention tiles.
// ---------------------------------------------------------------------------

typedef __bf16  v16bf __attribute__((ext_vector_type(16)));
typedef __bf16  v2bf  __attribute__((ext_vector_type(2)));
typedef float   v8f   __attribute__((ext_vector_type(8)));

#define Bsz 2
#define Tsz 2048
#define Psz 1024
#define Hsz 16
#define Dsz 128
#define Csz 2048
#define Lsz (Psz + Tsz)   /* 3072 total keys */
#define Msz (Bsz * Tsz)   /* 4096 rows */

union FragU { v16bf v; uint4 q[2]; };

// fp32 -> bf16 via hardware convert (v_cvt_pk_bf16_f32 class ops).
__device__ __forceinline__ unsigned short f2bf(float f) {
  return __builtin_bit_cast(unsigned short, (__bf16)f);
}
__device__ __forceinline__ unsigned int packbf2(float lo, float hi) {
  v2bf v; v[0] = (__bf16)lo; v[1] = (__bf16)hi;
  return __builtin_bit_cast(unsigned int, v);
}

// Async DMA: global -> LDS, 16 bytes per lane, tracked by ASYNCcnt.
__device__ __forceinline__ void async_copy16(const unsigned short* g, unsigned short* l) {
  const unsigned lds_off = (unsigned)(unsigned long long)l;   // addr[31:0] == LDS offset
  asm volatile("global_load_async_to_lds_b128 %0, %1, off"
               :: "v"(lds_off), "v"((unsigned long long)g) : "memory");
}
__device__ __forceinline__ void wait_async_all() {
  asm volatile("s_wait_asynccnt 0x0" ::: "memory");
}

// A-matrix fragment (16x32 bf16, M x K).  ISA 7.12.2: lane holds row M=lane&15;
// VGPR0-3 cover K = half*8 + 0..7, VGPR4-7 cover K = 16 + half*8 + 0..7.
__device__ __forceinline__ v16bf load_frag_a(const unsigned short* base,
                                             int m0, int k0, int ldm, int lane) {
  const int m  = m0 + (lane & 15);
  const int hf = (lane >> 4) & 1;
  const unsigned short* p = base + m * ldm + k0;
  FragU u;
  u.q[0] = *(const uint4*)(p + hf * 8);
  u.q[1] = *(const uint4*)(p + 16 + hf * 8);
  return u.v;
}

// B-matrix fragment (32x16 bf16, K x N) from an LDS tile stored transposed as
// [n][k].  ISA: lane holds column N=lane&15; elems 0-7 K=half*8+0..7, 8-15 K=16+half*8+0..7.
__device__ __forceinline__ v16bf load_frag_b(const unsigned short* baseT,
                                             int n0, int k0, int ldn, int lane) {
  const int n  = n0 + (lane & 15);
  const int hf = (lane >> 4) & 1;
  const unsigned short* p = baseT + n * ldn + k0;
  FragU u;
  u.q[0] = *(const uint4*)(p + hf * 8);
  u.q[1] = *(const uint4*)(p + 16 + hf * 8);
  return u.v;
}

// ---------------------------------------------------------------------------
// Generic fp32-in / fp32-out GEMM, bf16 WMMA core.  C[M,N] = A[M,K] * B[K,N].
// 128x128 tile / 256-thread block, software-pipelined: tile kt+1 is fetched
// into registers while WMMAs consume tile kt from LDS.
// ---------------------------------------------------------------------------
struct TileRegs { float4 fa[4]; float4 fb[4]; };

__device__ __forceinline__ void gemm_load_tiles(const float* __restrict__ A,
                                                const float* __restrict__ Bm,
                                                int K, int N, int rowBase, int colBase,
                                                int k0, int tid, TileRegs& t) {
#pragma unroll
  for (int i = 0; i < 4; ++i) {
    const int idx = tid + i * 256;
    t.fa[i] = *(const float4*)(A + (size_t)(rowBase + (idx >> 3)) * K + k0 + ((idx & 7) << 2));
  }
#pragma unroll
  for (int i = 0; i < 4; ++i) {
    const int idx = tid + i * 256;
    t.fb[i] = *(const float4*)(Bm + (size_t)(k0 + (idx >> 5)) * N + colBase + ((idx & 31) << 2));
  }
}

__device__ __forceinline__ void gemm_store_tiles(unsigned short* sA, unsigned short* sB,
                                                 int tid, const TileRegs& t) {
#pragma unroll
  for (int i = 0; i < 4; ++i) {            // A: [m][k], packed 2-at-a-time
    const int idx = tid + i * 256;
    const int r = idx >> 3, c = (idx & 7) << 2;
    unsigned int* d = (unsigned int*)(sA + r * 32 + c);
    d[0] = packbf2(t.fa[i].x, t.fa[i].y);
    d[1] = packbf2(t.fa[i].z, t.fa[i].w);
  }
#pragma unroll
  for (int i = 0; i < 4; ++i) {            // B: transposed [n][k]
    const int idx = tid + i * 256;
    const int kr = idx >> 5, n4 = (idx & 31) << 2;
    sB[(n4 + 0) * 32 + kr] = f2bf(t.fb[i].x);
    sB[(n4 + 1) * 32 + kr] = f2bf(t.fb[i].y);
    sB[(n4 + 2) * 32 + kr] = f2bf(t.fb[i].z);
    sB[(n4 + 3) * 32 + kr] = f2bf(t.fb[i].w);
  }
}

__global__ __launch_bounds__(256)
void mha_gemm_bf16_wmma(const float* __restrict__ A, const float* __restrict__ Bm,
                        float* __restrict__ Cm, int M, int N, int K) {
  __shared__ __align__(16) unsigned short sA[128 * 32];
  __shared__ __align__(16) unsigned short sB[128 * 32];

  const int tid  = threadIdx.x;
  const int lane = tid & 31;
  const int wave = tid >> 5;
  const int wM = wave >> 1, wN = wave & 1;
  const int rowBase = blockIdx.y * 128;
  const int colBase = blockIdx.x * 128;

  v8f acc[2][4];
#pragma unroll
  for (int i = 0; i < 2; ++i)
#pragma unroll
    for (int j = 0; j < 4; ++j)
#pragma unroll
      for (int r = 0; r < 8; ++r) acc[i][j][r] = 0.0f;

  const int nK = K >> 5;
  TileRegs t;
  gemm_load_tiles(A, Bm, K, N, rowBase, colBase, 0, tid, t);
  gemm_store_tiles(sA, sB, tid, t);
  __syncthreads();

  for (int kt = 0; kt < nK; ++kt) {
    const bool more = (kt + 1 < nK);
    if (more) gemm_load_tiles(A, Bm, K, N, rowBase, colBase, (kt + 1) << 5, tid, t);
    if (kt + 2 < nK)   // global_prefetch_b8 a little further ahead
      __builtin_prefetch(A + (size_t)(rowBase + (tid >> 1)) * K + ((kt + 2) << 5) + (tid & 1) * 16, 0, 1);

    const v16bf a0 = load_frag_a(sA, wM * 32,      0, 32, lane);
    const v16bf a1 = load_frag_a(sA, wM * 32 + 16, 0, 32, lane);
#pragma unroll
    for (int nf = 0; nf < 4; ++nf) {
      const v16bf b = load_frag_b(sB, wN * 64 + nf * 16, 0, 32, lane);
      acc[0][nf] = __builtin_amdgcn_wmma_f32_16x16x32_bf16(false, a0, false, b,
                                                           (short)0, acc[0][nf], false, false);
      acc[1][nf] = __builtin_amdgcn_wmma_f32_16x16x32_bf16(false, a1, false, b,
                                                           (short)0, acc[1][nf], false, false);
    }
    __syncthreads();
    if (more) gemm_store_tiles(sA, sB, tid, t);
    __syncthreads();
  }

  const int hf = lane >> 4, ln = lane & 15;
#pragma unroll
  for (int mf = 0; mf < 2; ++mf)
#pragma unroll
    for (int nf = 0; nf < 4; ++nf)
#pragma unroll
      for (int r = 0; r < 8; ++r) {
        const int m = rowBase + wM * 32 + mf * 16 + r + hf * 8;
        const int n = colBase + wN * 64 + nf * 16 + ln;
        Cm[(size_t)m * N + n] = acc[mf][nf][r];
      }
}

// ---------------------------------------------------------------------------
// past_k/past_v fp32 [B,H,P,D] -> bf16 KV cache [B,H,L,D] positions 0..P-1
// ---------------------------------------------------------------------------
__global__ __launch_bounds__(256)
void mha_past_kv(const float* __restrict__ pk, const float* __restrict__ pv,
                 unsigned short* __restrict__ Kb, unsigned short* __restrict__ Vb) {
  const int gid = blockIdx.x * 256 + threadIdx.x;      // 2^22 total
  const int d = gid & 127;
  const int p = (gid >> 7) & 1023;
  const int h = (gid >> 17) & 15;
  const int b = gid >> 21;
  const size_t dst = (((size_t)(b * Hsz + h)) * Lsz + p) * Dsz + d;
  Kb[dst] = f2bf(pk[gid]);
  Vb[dst] = f2bf(pv[gid]);
}

// ---------------------------------------------------------------------------
// RoPE (GPT-NeoX rotate-half, positions P..P+T-1) + split-heads scatter.
// ---------------------------------------------------------------------------
__global__ __launch_bounds__(256)
void mha_rope_scatter(const float* __restrict__ qkv, unsigned short* __restrict__ Qb,
                      unsigned short* __restrict__ Kb, unsigned short* __restrict__ Vb) {
  const int gid = blockIdx.x * 256 + threadIdx.x;      // 2^23 total
  const int d = gid & 127;
  const int t = (gid >> 7) & 2047;
  const int h = (gid >> 18) & 15;
  const int b = gid >> 22;
  const size_t base = ((size_t)(b * Tsz + t)) * (3 * Csz) + h * Dsz;
  const float qv = qkv[base + d];
  const float kv = qkv[base + Csz + d];
  const float vv = qkv[base + 2 * Csz + d];
  const float qp = qkv[base + (d ^ 64)];
  const float kp = qkv[base + Csz + (d ^ 64)];
  const int i = d & 63;
  const float invf = __expf(-(float)i * (9.210340371976184f / 64.0f)); // theta^(-i/64)
  const float ang = (float)(Psz + t) * invf;
  float s, c;
  __sincosf(ang, &s, &c);
  const float sgn = (d < 64) ? -1.0f : 1.0f;
  const size_t qi = (((size_t)(b * Hsz + h)) * Tsz + t) * Dsz + d;
  const size_t ki = (((size_t)(b * Hsz + h)) * Lsz + Psz + t) * Dsz + d;
  Qb[qi] = f2bf(qv * c + sgn * qp * s);
  Kb[ki] = f2bf(kv * c + sgn * kp * s);
  Vb[ki] = f2bf(vv);
}

// ---------------------------------------------------------------------------
// Flash attention.  Block = (128 q-rows, one (b,h)); 8 waves x 16 q-rows.
// Q + K tiles stream via global_load_async_to_lds_b128 (ASYNCcnt); K is
// double-buffered so the next 32-key block DMA overlaps WMMA + softmax.
// V is register-pipelined one block ahead (needs [d][key] transpose in LDS).
// ---------------------------------------------------------------------------
__global__ __launch_bounds__(256)
void mha_flash_attn(const unsigned short* __restrict__ Q,
                    const unsigned short* __restrict__ Kc,
                    const unsigned short* __restrict__ Vc,
                    float* __restrict__ Out) {
  __shared__ __align__(16) unsigned short sQ[128 * 128];     // [qrow][d]  32 KB
  __shared__ __align__(16) unsigned short sK[2][32 * 128];   // [key][d]   16 KB
  __shared__ __align__(16) unsigned short sV[128 * 32];      // [d][key]    8 KB
  __shared__ __align__(16) unsigned short sP[8 * 16 * 32];   // per-wave P  8 KB

  const int tid  = threadIdx.x;
  const int lane = tid & 31;
  const int wave = tid >> 5;
  const int hf = lane >> 4, ln = lane & 15;
  const int qBase = blockIdx.x * 128;
  const int h = blockIdx.y;
  const int b = blockIdx.z;

  const unsigned short* Qp = Q  + (((size_t)(b * Hsz + h)) * Tsz + qBase) * Dsz;
  const unsigned short* Kp = Kc + ((size_t)(b * Hsz + h)) * Lsz * Dsz;
  const unsigned short* Vp = Vc + ((size_t)(b * Hsz + h)) * Lsz * Dsz;

  // ---- prologue: async Q tile + K block 0; V block 0 through registers ----
#pragma unroll
  for (int i = 0; i < 8; ++i) {
    const int idx = tid + i * 256;
    async_copy16(Qp + idx * 8, sQ + idx * 8);
  }
#pragma unroll
  for (int i = 0; i < 2; ++i) {
    const int idx = tid + i * 256;
    async_copy16(Kp + idx * 8, sK[0] + idx * 8);
  }
  const int vkk = tid >> 3;             // key 0..31
  const int vd0 = (tid & 7) * 16;       // d 0..112
  union U8 { uint4 q; unsigned short e[8]; } v0, v1;
  v0.q = *(const uint4*)(Vp + vkk * Dsz + vd0);
  v1.q = *(const uint4*)(Vp + vkk * Dsz + vd0 + 8);
#pragma unroll
  for (int j = 0; j < 8; ++j) sV[(vd0 + j) * 32 + vkk] = v0.e[j];
#pragma unroll
  for (int j = 0; j < 8; ++j) sV[(vd0 + 8 + j) * 32 + vkk] = v1.e[j];
  wait_async_all();
  __syncthreads();

  float Mr[8], Lr[8];
  v8f o[8];
#pragma unroll
  for (int r = 0; r < 8; ++r) { Mr[r] = -1.0e30f; Lr[r] = 0.0f; }
#pragma unroll
  for (int f = 0; f < 8; ++f)
#pragma unroll
    for (int r = 0; r < 8; ++r) o[f][r] = 0.0f;

  const float scale = 0.08838834764831845f;   // 1/sqrt(128)
  unsigned short* myP = sP + wave * (16 * 32);
  const int NKB = Lsz / 32;

  for (int kb = 0; kb < NKB; ++kb) {
    const int cur = kb & 1;
    if (kb + 1 < NKB) {                 // stream next K block (DMA) + V regs
      const unsigned short* kp = Kp + (size_t)(kb + 1) * 32 * Dsz;
#pragma unroll
      for (int i = 0; i < 2; ++i) {
        const int idx = tid + i * 256;
        async_copy16(kp + idx * 8, sK[1 - cur] + idx * 8);
      }
      const unsigned short* vp = Vp + (size_t)(kb + 1) * 32 * Dsz;
      v0.q = *(const uint4*)(vp + vkk * Dsz + vd0);
      v1.q = *(const uint4*)(vp + vkk * Dsz + vd0 + 8);
    }

    // S = Q * K^T : 16 q-rows x 32 keys per wave, contracted over d=128.
    v8f s0, s1;
#pragma unroll
    for (int r = 0; r < 8; ++r) { s0[r] = 0.0f; s1[r] = 0.0f; }
#pragma unroll
    for (int dc = 0; dc < 4; ++dc) {
      const v16bf a  = load_frag_a(sQ, wave * 16, dc * 32, 128, lane);
      const v16bf b0 = load_frag_b(sK[cur], 0,  dc * 32, 128, lane);
      const v16bf b1 = load_frag_b(sK[cur], 16, dc * 32, 128, lane);
      s0 = __builtin_amdgcn_wmma_f32_16x16x32_bf16(false, a, false, b0, (short)0, s0, false, false);
      s1 = __builtin_amdgcn_wmma_f32_16x16x32_bf16(false, a, false, b1, (short)0, s1, false, false);
    }

    // Online softmax (row = r + hf*8 lives in the 16 lanes of this half).
#pragma unroll
    for (int r = 0; r < 8; ++r) {
      const float x0 = s0[r] * scale;
      const float x1 = s1[r] * scale;
      float mx = fmaxf(x0, x1);
#pragma unroll
      for (int off = 8; off >= 1; off >>= 1) mx = fmaxf(mx, __shfl_xor(mx, off, 32));
      const float mnew  = fmaxf(Mr[r], mx);
      const float alpha = __expf(Mr[r] - mnew);
      const float p0 = __expf(x0 - mnew);
      const float p1 = __expf(x1 - mnew);
      float rs = p0 + p1;
#pragma unroll
      for (int off = 8; off >= 1; off >>= 1) rs += __shfl_xor(rs, off, 32);
      Lr[r] = Lr[r] * alpha + rs;
      Mr[r] = mnew;
#pragma unroll
      for (int f = 0; f < 8; ++f) o[f][r] *= alpha;
      const int row = r + hf * 8;
      myP[row * 32 + ln]      = f2bf(p0);
      myP[row * 32 + ln + 16] = f2bf(p1);
    }
    asm volatile("s_wait_dscnt 0x0" ::: "memory");   // wave-local P visibility

    // O += P(16x32) * V(32x128)
    const v16bf pf = load_frag_a(myP, 0, 0, 32, lane);
#pragma unroll
    for (int nf = 0; nf < 8; ++nf) {
      const v16bf vf = load_frag_b(sV, nf * 16, 0, 32, lane);
      o[nf] = __builtin_amdgcn_wmma_f32_16x16x32_bf16(false, pf, false, vf, (short)0, o[nf], false, false);
    }

    __syncthreads();                    // all waves done with sV / sK[cur]
    if (kb + 1 < NKB) {                 // commit next V block (transposed)
#pragma unroll
      for (int j = 0; j < 8; ++j) sV[(vd0 + j) * 32 + vkk] = v0.e[j];
#pragma unroll
      for (int j = 0; j < 8; ++j) sV[(vd0 + 8 + j) * 32 + vkk] = v1.e[j];
    }
    wait_async_all();                   // next K block resident
    __syncthreads();
  }

  // Normalize and write merged-head output [B,T,C].
#pragma unroll
  for (int r = 0; r < 8; ++r) {
    const float inv = 1.0f / Lr[r];
    const int m = qBase + wave * 16 + r + hf * 8;
    float* op = Out + ((size_t)b * Tsz + m) * Csz + h * Dsz;
#pragma unroll
    for (int nf = 0; nf < 8; ++nf) op[nf * 16 + ln] = o[nf][r] * inv;
  }
}

// ---------------------------------------------------------------------------
extern "C" void kernel_launch(void* const* d_in, const int* in_sizes, int n_in,
                              void* d_out, int out_size, void* d_ws, size_t ws_size,
                              hipStream_t stream) {
  (void)in_sizes; (void)n_in; (void)out_size; (void)ws_size;
  const float* x     = (const float*)d_in[0];
  const float* w_qkv = (const float*)d_in[1];
  const float* w_out = (const float*)d_in[2];
  const float* pk    = (const float*)d_in[3];
  const float* pv    = (const float*)d_in[4];
  float* out = (float*)d_out;

  // Workspace (~168 MB):
  //   [0)          qkv fp32 [B,T,3C] (100.7 MB) -- ALIASED by attnO afterwards
  //   [+100663296) Qb bf16 [B,H,T,D] (16.8 MB)
  //   [+117440512) Kb bf16 [B,H,L,D] (25.2 MB)
  //   [+142606336) Vb bf16 [B,H,L,D] (25.2 MB)
  char* ws = (char*)d_ws;
  float* qkv   = (float*)ws;
  float* attnO = (float*)ws;   // alias: qkv dead once rope_scatter completes
  size_t off = (size_t)Bsz * Tsz * 3 * Csz * sizeof(float);
  unsigned short* Qb = (unsigned short*)(ws + off); off += (size_t)Bsz * Hsz * Tsz * Dsz * 2;
  unsigned short* Kb = (unsigned short*)(ws + off); off += (size_t)Bsz * Hsz * Lsz * Dsz * 2;
  unsigned short* Vb = (unsigned short*)(ws + off);

  mha_gemm_bf16_wmma<<<dim3((3 * Csz) / 128, Msz / 128), 256, 0, stream>>>(
      x, w_qkv, qkv, Msz, 3 * Csz, Csz);
  mha_past_kv<<<(Bsz * Hsz * Psz * Dsz) / 256, 256, 0, stream>>>(pk, pv, Kb, Vb);
  mha_rope_scatter<<<(Bsz * Hsz * Tsz * Dsz) / 256, 256, 0, stream>>>(qkv, Qb, Kb, Vb);
  mha_flash_attn<<<dim3(Tsz / 128, Hsz, Bsz), 256, 0, stream>>>(Qb, Kb, Vb, attnO);
  mha_gemm_bf16_wmma<<<dim3(Csz / 128, Msz / 128), 256, 0, stream>>>(
      attnO, w_out, out, Msz, Csz, Csz);
}
